// SlidingWindowAttention_70609262346604
// MI455X (gfx1250) — matrix operand order repaired and checked by reading, not compiled, not run on hardware
//
#include <hip/hip_runtime.h>
#include <hip/hip_bf16.h>

// ---------------------------------------------------------------------------
// Types for CDNA5 WMMA (wave32): v_wmma_f32_16x16x32_f16
// ---------------------------------------------------------------------------
typedef __attribute__((ext_vector_type(16))) _Float16 v16h;
typedef __attribute__((ext_vector_type(8)))  _Float16 v8h;
typedef __attribute__((ext_vector_type(8)))  float    v8f;

#define BATCH   2
#define SEQ     4096
#define DMODEL  1024
#define NHEADS  16
#define HDIM    64
#define WIN     512

__device__ __forceinline__ v8f wmma_f16(v16h a, v16h b, v8f c) {
  return __builtin_amdgcn_wmma_f32_16x16x32_f16(false, a, false, b, (short)0, c,
                                                false, false);
}

// A-operand fragment (16x32 f16, M rows x K): lane l holds row (l&15);
// elements 0..7 = K (l>>4)*8 .. +7 ; elements 8..15 = K 16+(l>>4)*8 .. +7
__device__ __forceinline__ v16h load_frag_a(const _Float16* base, int ld, int lane) {
  const int r  = lane & 15;
  const int kh = (lane >> 4) << 3;         // 0 or 8
  const _Float16* p = base + (long)r * ld + kh;
  v8h lo = *(const v8h*)(p);
  v8h hi = *(const v8h*)(p + 16);
  v16h f;
#pragma unroll
  for (int i = 0; i < 8; ++i) { f[i] = lo[i]; f[i + 8] = hi[i]; }
  return f;
}

// B-operand fragment (32x16 f16, K x N) loaded from memory stored [N, K]:
// lane l holds column n=(l&15); elements 0..7 = K (l>>4)*8 .. +7 ; elements 8..15 = K 16+(l>>4)*8 .. +7
__device__ __forceinline__ v16h load_frag_b(const _Float16* base, int ld, int lane) {
  const int n  = lane & 15;
  const int kh = (lane >> 4) << 3;         // 0 or 8
  const _Float16* p = base + (long)n * ld + kh;
  v8h lo = *(const v8h*)(p);
  v8h hi = *(const v8h*)(p + 16);
  v16h f;
#pragma unroll
  for (int i = 0; i < 8; ++i) { f[i] = lo[i]; f[i + 8] = hi[i]; }
  return f;
}

// ---------------------------------------------------------------------------
// fp32 -> f16 weight conversion
// ---------------------------------------------------------------------------
__global__ void f32_to_f16_kernel(const float* __restrict__ in,
                                  _Float16* __restrict__ out, int n) {
  for (int i = blockIdx.x * blockDim.x + threadIdx.x; i < n;
       i += gridDim.x * blockDim.x)
    out[i] = (_Float16)in[i];
}

// ---------------------------------------------------------------------------
// LayerNorm (fp32 in) -> f16 out.  One block (256 thr) per row, D=1024.
// ---------------------------------------------------------------------------
__global__ __launch_bounds__(256)
void ln_f16_kernel(const float* __restrict__ x, const float* __restrict__ g,
                   const float* __restrict__ bta, _Float16* __restrict__ out) {
  __shared__ float red0[8], red1[8];
  const int row = blockIdx.x;
  const float* xr = x + (long)row * DMODEL;
  float vals[4];
  float lsum = 0.f, lsq = 0.f;
#pragma unroll
  for (int i = 0; i < 4; ++i) {
    float t = xr[threadIdx.x + i * 256];
    vals[i] = t; lsum += t; lsq += t * t;
  }
#pragma unroll
  for (int off = 16; off > 0; off >>= 1) {
    lsum += __shfl_xor(lsum, off);
    lsq  += __shfl_xor(lsq,  off);
  }
  const int w = threadIdx.x >> 5;
  if ((threadIdx.x & 31) == 0) { red0[w] = lsum; red1[w] = lsq; }
  __syncthreads();
  float s = 0.f, q = 0.f;
#pragma unroll
  for (int i = 0; i < 8; ++i) { s += red0[i]; q += red1[i]; }
  const float mean = s * (1.0f / DMODEL);
  const float var  = q * (1.0f / DMODEL) - mean * mean;
  const float rstd = rsqrtf(var + 1e-5f);
#pragma unroll
  for (int i = 0; i < 4; ++i) {
    int c = threadIdx.x + i * 256;
    out[(long)row * DMODEL + c] =
        (_Float16)((vals[i] - mean) * rstd * g[c] + bta[c]);
  }
}

// ---------------------------------------------------------------------------
// Generic NT GEMM:  C[M,N] = A[M,K] * B[N,K]^T  (f16 operands, f32 acc)
// Block = 128 thr = 4 waves (2x2), each wave owns a 64x64 tile (16 WMMA/k-step).
// Epilogues:
//   0: QKV split -> q(scaled)/k f16 [B,H,S,hd], v f16 transposed [B,H,hd,S]
//      (v uses packed 16B stores: each lane holds 8 consecutive s for fixed d)
//   1: residual:  outf = xin + acc                    (fp32)
//   2: bias+GELU(exact erf) -> f16
//   3: final:     outf = xin + acc + bias             (fp32)
// ---------------------------------------------------------------------------
struct GemmArgs {
  const _Float16* A; const _Float16* B;
  int M, N, K;
  const float* xin; const float* bias;
  float* outf; _Float16* outh;
  _Float16* q; _Float16* k; _Float16* vt;
  float scale;
};

template <int EPI>
__global__ __launch_bounds__(128)
void gemm_nt_kernel(GemmArgs ga) {
  const int lane = threadIdx.x & 31;
  const int wave = threadIdx.x >> 5;
  const int wm = wave >> 1, wn = wave & 1;
  const int m0 = blockIdx.y * 128 + wm * 64;
  const int n0 = blockIdx.x * 128 + wn * 64;

  v8f acc[4][4];
  const v8f vzero = {0.f, 0.f, 0.f, 0.f, 0.f, 0.f, 0.f, 0.f};
#pragma unroll
  for (int i = 0; i < 4; ++i)
#pragma unroll
    for (int j = 0; j < 4; ++j) acc[i][j] = vzero;

  const _Float16* Abase = ga.A + (long)m0 * ga.K;
  const _Float16* Bbase = ga.B + (long)n0 * ga.K;

  for (int k0 = 0; k0 < ga.K; k0 += 32) {
    // warm L2->L0 for the panel two k-steps ahead (global_prefetch_b8)
    if (k0 + 64 < ga.K) {
      __builtin_prefetch(Abase + (long)lane * ga.K + k0 + 64, 0, 1);
      __builtin_prefetch(Bbase + (long)lane * ga.K + k0 + 64, 0, 1);
    }
    v16h af[4], bf[4];
#pragma unroll
    for (int i = 0; i < 4; ++i)
      af[i] = load_frag_a(Abase + (long)(i * 16) * ga.K + k0, ga.K, lane);
#pragma unroll
    for (int j = 0; j < 4; ++j)
      bf[j] = load_frag_b(Bbase + (long)(j * 16) * ga.K + k0, ga.K, lane);
#pragma unroll
    for (int i = 0; i < 4; ++i)
#pragma unroll
      for (int j = 0; j < 4; ++j)
        acc[i][j] = wmma_f16(af[i], bf[j], acc[i][j]);
  }

  // C/D layout: VGPR r, lane l -> M = (l>>4)*8 + r , N = l&15
  const int rl = lane & 15;
  const int hh = lane >> 4;
#pragma unroll
  for (int i = 0; i < 4; ++i) {
#pragma unroll
    for (int j = 0; j < 4; ++j) {
      if (EPI == 0) {
        const int nt0 = n0 + j * 16;            // tile n base (16-wide)
        const int sel = nt0 >> 10;              // 0:q 1:k 2:v (tile-uniform)
        const int mt0 = m0 + i * 16;
        const int b = mt0 >> 12;                // mt0 / SEQ
        if (sel == 2) {
          // V transposed: lane holds d = rl fixed, s = sbase..sbase+7 contig
          const int nn = (nt0 & (DMODEL - 1)) + rl;
          const int h = nn >> 6, d = nn & 63;
          const long hb = (long)(b * NHEADS + h);
          const int sbase = (mt0 & (SEQ - 1)) + hh * 8;
          v8h pack;
#pragma unroll
          for (int r = 0; r < 8; ++r) pack[r] = (_Float16)acc[i][j][r];
          *(v8h*)(ga.vt + (hb * HDIM + d) * (long)SEQ + sbase) = pack;
        } else {
#pragma unroll
          for (int r = 0; r < 8; ++r) {
            const int s = (mt0 & (SEQ - 1)) + hh * 8 + r;
            const int nn = (nt0 & (DMODEL - 1)) + rl;
            const int h = nn >> 6, d = nn & 63;
            const long hb = (long)(b * NHEADS + h);
            const float v = acc[i][j][r];
            if (sel == 0)
              ga.q[(hb * SEQ + s) * HDIM + d] = (_Float16)(v * ga.scale);
            else
              ga.k[(hb * SEQ + s) * HDIM + d] = (_Float16)v;
          }
        }
      } else {
#pragma unroll
        for (int r = 0; r < 8; ++r) {
          const int m = m0 + i * 16 + hh * 8 + r;
          const int n = n0 + j * 16 + rl;
          const float v = acc[i][j][r];
          if (EPI == 1) {
            const long idx = (long)m * ga.N + n;
            ga.outf[idx] = ga.xin[idx] + v;
          } else if (EPI == 2) {
            const float t = v + ga.bias[n];
            const float gl = 0.5f * t * (1.0f + erff(t * 0.70710678118654752f));
            ga.outh[(long)m * ga.N + n] = (_Float16)gl;
          } else {
            const long idx = (long)m * ga.N + n;
            ga.outf[idx] = ga.xin[idx] + v + ga.bias[n];
          }
        }
      }
    }
  }
}

// ---------------------------------------------------------------------------
// Sliding-window attention, f16 WMMA + online softmax.
// Grid: B*H*C*(W/64) blocks, 128 thr = 4 waves; each wave = 16 queries.
// q pre-scaled by 1/sqrt(hd).  Keys j in [0,2W) concat space,
// valid iff i < j <= i+W (and j>=W when c==0).  Only tiles intersecting
// [i0+1, i0+15+W] are processed (~2x work saving).
// ---------------------------------------------------------------------------
__global__ __launch_bounds__(128)
void attn_kernel(const _Float16* __restrict__ q, const _Float16* __restrict__ k,
                 const _Float16* __restrict__ vt, _Float16* __restrict__ attn) {
  __shared__ _Float16 plds[4][16 * 32];     // per-wave P tile (A-layout source)

  const int lane = threadIdx.x & 31;
  const int wave = threadIdx.x >> 5;
  const int bid = blockIdx.x;
  const int qb = bid & 7;                    // 64-query group in chunk
  const int c  = (bid >> 3) & 7;             // chunk
  const int h  = (bid >> 6) & 15;
  const int b  = bid >> 10;

  const int i0 = qb * 64 + wave * 16;        // local query base in [0,512)
  const long head = (long)(b * NHEADS + h);
  const _Float16* qh = q  + head * (long)SEQ * HDIM;
  const _Float16* kh = k  + head * (long)SEQ * HDIM;
  const _Float16* vh = vt + head * (long)HDIM * SEQ;

  const int sq0 = c * WIN + i0;              // absolute query base
  const v16h aq0 = load_frag_a(qh + (long)sq0 * HDIM + 0,  HDIM, lane);
  const v16h aq1 = load_frag_a(qh + (long)sq0 * HDIM + 32, HDIM, lane);

  const v8f vzero = {0.f, 0.f, 0.f, 0.f, 0.f, 0.f, 0.f, 0.f};
  v8f acc[4];
#pragma unroll
  for (int nt = 0; nt < 4; ++nt) acc[nt] = vzero;
  float rmax[8], rsum[8];
#pragma unroll
  for (int r = 0; r < 8; ++r) { rmax[r] = -1e30f; rsum[r] = 0.f; }

  const int rl = lane & 15;
  const int hh = lane >> 4;

  const int jstart = (c == 0) ? WIN : ((i0 + 1) & ~31);
  const int jend   = i0 + 15 + WIN;          // inclusive

  for (int j0 = jstart; j0 <= jend; j0 += 32) {
    const int ka = (c - 1) * WIN + j0;       // absolute key base (>= 0)
    // --- scores: two 16x16 tiles, K=64 in two WMMA each ---
    v8f s0 = vzero, s1 = vzero;
    v16h kf;
    kf = load_frag_b(kh + (long)ka * HDIM + 0,         HDIM, lane); s0 = wmma_f16(aq0, kf, s0);
    kf = load_frag_b(kh + (long)ka * HDIM + 32,        HDIM, lane); s0 = wmma_f16(aq1, kf, s0);
    kf = load_frag_b(kh + (long)(ka + 16) * HDIM + 0,  HDIM, lane); s1 = wmma_f16(aq0, kf, s1);
    kf = load_frag_b(kh + (long)(ka + 16) * HDIM + 32, HDIM, lane); s1 = wmma_f16(aq1, kf, s1);

    // --- mask + online softmax (row = (hh*8+r), col = rl across 16 lanes) ---
    float corr[8];
#pragma unroll
    for (int r = 0; r < 8; ++r) {
      const int iq = i0 + hh * 8 + r;        // local query index
      const int jA = j0 + rl, jB = jA + 16;
      float sa = (jA > iq && jA <= iq + WIN) ? s0[r] : -1e30f;
      float sb = (jB > iq && jB <= iq + WIN) ? s1[r] : -1e30f;
      float mx = fmaxf(sa, sb);
#pragma unroll
      for (int off = 1; off < 16; off <<= 1) mx = fmaxf(mx, __shfl_xor(mx, off));
      const float nm = fmaxf(rmax[r], mx);
      const float cr = __expf(rmax[r] - nm);
      rmax[r] = nm;
      const float pa = __expf(sa - nm), pb = __expf(sb - nm);
      float ps = pa + pb;
#pragma unroll
      for (int off = 1; off < 16; off <<= 1) ps += __shfl_xor(ps, off);
      rsum[r] = rsum[r] * cr + ps;
      corr[r] = cr;
      const int m = hh * 8 + r;
      plds[wave][m * 32 + rl]      = (_Float16)pa;
      plds[wave][m * 32 + 16 + rl] = (_Float16)pb;
    }
#pragma unroll
    for (int nt = 0; nt < 4; ++nt)
#pragma unroll
      for (int r = 0; r < 8; ++r) acc[nt][r] *= corr[r];

    // wave-private LDS relayout: C-layout -> A-layout
    asm volatile("s_wait_dscnt 0" ::: "memory");
    const v16h pf = load_frag_a(&plds[wave][0], 32, lane);

    // --- P (16x32) x V^T tiles (32 keys x 16 hd), hd = 4 tiles ---
#pragma unroll
    for (int nt = 0; nt < 4; ++nt) {
      const v16h vf = load_frag_b(vh + (long)(nt * 16) * SEQ + ka, SEQ, lane);
      acc[nt] = wmma_f16(pf, vf, acc[nt]);
    }
  }

  // --- finalize: out f16 [B, S, H*hd] ---
  float rinv[8];
#pragma unroll
  for (int r = 0; r < 8; ++r) rinv[r] = 1.0f / rsum[r];
#pragma unroll
  for (int nt = 0; nt < 4; ++nt) {
#pragma unroll
    for (int r = 0; r < 8; ++r) {
      const int m = hh * 8 + r;
      const int sq = sq0 + m;
      const int d = nt * 16 + rl;
      attn[((long)(b * SEQ + sq)) * DMODEL + h * HDIM + d] =
          (_Float16)(acc[nt][r] * rinv[r]);
    }
  }
}

// ---------------------------------------------------------------------------
// Launch: LN1 -> QKV -> attention -> w_o(+resid) -> LN2 -> MLP1(GELU) -> MLP2
// ---------------------------------------------------------------------------
extern "C" void kernel_launch(void* const* d_in, const int* in_sizes, int n_in,
                              void* d_out, int out_size, void* d_ws,
                              size_t ws_size, hipStream_t stream) {
  const float* x      = (const float*)d_in[0];
  const float* w_qkv  = (const float*)d_in[1];
  const float* w_o    = (const float*)d_in[2];
  const float* ln1_g  = (const float*)d_in[3];
  const float* ln1_b  = (const float*)d_in[4];
  const float* ln2_g  = (const float*)d_in[5];
  const float* ln2_b  = (const float*)d_in[6];
  const float* mlp_w1 = (const float*)d_in[7];
  const float* mlp_b1 = (const float*)d_in[8];
  const float* mlp_w2 = (const float*)d_in[9];
  const float* mlp_b2 = (const float*)d_in[10];
  float* out = (float*)d_out;

  char* ws = (char*)d_ws;
  const long MB = 1024 * 1024;
  _Float16* wqkv16 = (_Float16*)(ws + 0 * MB);    // 6 MB
  _Float16* wo16   = (_Float16*)(ws + 6 * MB);    // 2 MB
  _Float16* w116   = (_Float16*)(ws + 8 * MB);    // 8 MB
  _Float16* w216   = (_Float16*)(ws + 16 * MB);   // 8 MB
  _Float16* buf1   = (_Float16*)(ws + 24 * MB);   // 16 MB: h1, later h2
  _Float16* qbuf   = (_Float16*)(ws + 40 * MB);   // 16 MB
  _Float16* kbuf   = (_Float16*)(ws + 56 * MB);   // 16 MB
  _Float16* vtbuf  = (_Float16*)(ws + 72 * MB);   // 16 MB
  _Float16* attnb  = (_Float16*)(ws + 88 * MB);   // 16 MB
  float*    x2     = (float*)   (ws + 104 * MB);  // 32 MB
  _Float16* hmid   = (_Float16*)(ws + 40 * MB);   // 64 MB reuse (q..attn dead)

  const int ROWS = BATCH * SEQ;                   // 8192

  // weights -> f16
  f32_to_f16_kernel<<<1024, 256, 0, stream>>>(w_qkv, wqkv16, 3 * DMODEL * DMODEL);
  f32_to_f16_kernel<<<1024, 256, 0, stream>>>(w_o, wo16, DMODEL * DMODEL);
  f32_to_f16_kernel<<<1024, 256, 0, stream>>>(mlp_w1, w116, 4 * DMODEL * DMODEL);
  f32_to_f16_kernel<<<1024, 256, 0, stream>>>(mlp_w2, w216, 4 * DMODEL * DMODEL);

  // LN1
  ln_f16_kernel<<<ROWS, 256, 0, stream>>>(x, ln1_g, ln1_b, buf1);

  // QKV GEMM + split epilogue
  {
    GemmArgs g{}; g.A = buf1; g.B = wqkv16;
    g.M = ROWS; g.N = 3 * DMODEL; g.K = DMODEL;
    g.q = qbuf; g.k = kbuf; g.vt = vtbuf; g.scale = 0.125f;  // 1/sqrt(64)
    gemm_nt_kernel<0><<<dim3(g.N / 128, g.M / 128), 128, 0, stream>>>(g);
  }

  // sliding-window attention
  attn_kernel<<<BATCH * NHEADS * (SEQ / WIN) * (WIN / 64), 128, 0, stream>>>(
      qbuf, kbuf, vtbuf, attnb);

  // w_o GEMM + residual -> x2 (fp32)
  {
    GemmArgs g{}; g.A = attnb; g.B = wo16;
    g.M = ROWS; g.N = DMODEL; g.K = DMODEL;
    g.xin = x; g.outf = x2;
    gemm_nt_kernel<1><<<dim3(g.N / 128, g.M / 128), 128, 0, stream>>>(g);
  }

  // LN2 -> h2 (reuse buf1)
  ln_f16_kernel<<<ROWS, 256, 0, stream>>>(x2, ln2_g, ln2_b, buf1);

  // MLP1 + bias + exact GELU -> hmid f16
  {
    GemmArgs g{}; g.A = buf1; g.B = w116;
    g.M = ROWS; g.N = 4 * DMODEL; g.K = DMODEL;
    g.bias = mlp_b1; g.outh = hmid;
    gemm_nt_kernel<2><<<dim3(g.N / 128, g.M / 128), 128, 0, stream>>>(g);
  }

  // MLP2 + bias + residual -> out fp32
  {
    GemmArgs g{}; g.A = hmid; g.B = w216;
    g.M = ROWS; g.N = DMODEL; g.K = 4 * DMODEL;
    g.xin = x2; g.bias = mlp_b2; g.outf = out;
    gemm_nt_kernel<3><<<dim3(g.N / 128, g.M / 128), 128, 0, stream>>>(g);
  }
}
